// OPTMLSTMCell_29832842838104
// MI455X (gfx1250) — matrix operand order repaired and checked by reading, not compiled, not run on hardware
//
#include <hip/hip_runtime.h>
#include <stdint.h>

#define B_SZ   16384
#define IN_SZ  10
#define H_SZ   1024
#define SIXH   (6 * H_SZ)
#define GD_ITERS 7
#define GD_LR  0.001f
#define GD_ROWS 64

typedef __attribute__((ext_vector_type(16))) __bf16 v16bf;
typedef __attribute__((ext_vector_type(8)))  float  v8f;

// ---------------------------------------------------------------------------
// fp32 -> bf16 (round to nearest even), stored as raw u16. n % 4 == 0.
// ---------------------------------------------------------------------------
__global__ void f2bf_kernel(const float* __restrict__ src,
                            unsigned short* __restrict__ dst, int n4) {
    int i = blockIdx.x * blockDim.x + threadIdx.x;
    if (i < n4) {
        const float4 v = *(const float4*)(src + (size_t)i * 4);
        ushort4 h;
        unsigned u;
        u = __float_as_uint(v.x); h.x = (unsigned short)((u + 0x7fffu + ((u >> 16) & 1u)) >> 16);
        u = __float_as_uint(v.y); h.y = (unsigned short)((u + 0x7fffu + ((u >> 16) & 1u)) >> 16);
        u = __float_as_uint(v.z); h.z = (unsigned short)((u + 0x7fffu + ((u >> 16) & 1u)) >> 16);
        u = __float_as_uint(v.w); h.w = (unsigned short)((u + 0x7fffu + ((u >> 16) & 1u)) >> 16);
        *(ushort4*)(dst + (size_t)i * 4) = h;
    }
}

// ---------------------------------------------------------------------------
// Fused gates kernel: pre_g = hidden @ Ug^T (bf16 WMMA, fp32 acc)
//                              + x @ Wg^T + bg   (scalar epilogue, K=10)
// then elementwise LSTM math; writes r_t blocks [f,i,ct,o,cn,hs] and c_new.
// One wave = 32x16 output (two 16x16 tiles) for all 4 gates; the 4 B operands
// per k-step feed 8 WMMAs.
// ---------------------------------------------------------------------------
__global__ __launch_bounds__(256)
void gates_kernel(const unsigned short* __restrict__ hbf,   // [B][H] bf16
                  const unsigned short* __restrict__ ubf,   // [4][H][H] bf16 (U row-major)
                  const float* __restrict__ x,              // [B][IN]
                  const float* __restrict__ c_in,           // [B][H]
                  const float* __restrict__ Wf, const float* __restrict__ bf_,
                  const float* __restrict__ Wi, const float* __restrict__ bi_,
                  const float* __restrict__ Wc, const float* __restrict__ bc_,
                  const float* __restrict__ Wo, const float* __restrict__ bo_,
                  float* __restrict__ rt,                   // [B][6H]
                  float* __restrict__ cnew_out)             // [B][H]
{
    const int lane   = threadIdx.x & 31;
    const int wave   = threadIdx.x >> 5;
    const int row0   = blockIdx.x * 32;                 // two 16-row tiles
    const int col0   = (blockIdx.y * 8 + wave) * 16;
    const int mlane  = lane & 15;
    const int hiHalf = lane >> 4;                       // 0: lanes 0-15, 1: lanes 16-31

    const int bcol = col0 + mlane;                      // B/C/D: column owned by this lane

    // A: lane reads 16 contiguous K {base..base+15}, base = hiHalf*16
    const __bf16* aBase0 = (const __bf16*)(hbf + (size_t)(row0 + mlane) * H_SZ) + hiHalf * 16;
    const __bf16* aBase1 = (const __bf16*)(hbf + (size_t)(row0 + 16 + mlane) * H_SZ) + hiHalf * 16;
    // B[k,n] = U[n,k]: lane reads 16 contiguous K from U row bcol, offset hiHalf*16
    const __bf16* bBase0 = (const __bf16*)(ubf + 0u * H_SZ * H_SZ + (size_t)bcol * H_SZ) + hiHalf * 16;
    const __bf16* bBase1 = (const __bf16*)(ubf + 1u * H_SZ * H_SZ + (size_t)bcol * H_SZ) + hiHalf * 16;
    const __bf16* bBase2 = (const __bf16*)(ubf + 2u * H_SZ * H_SZ + (size_t)bcol * H_SZ) + hiHalf * 16;
    const __bf16* bBase3 = (const __bf16*)(ubf + 3u * H_SZ * H_SZ + (size_t)bcol * H_SZ) + hiHalf * 16;

    v8f acc0[2] = {{}, {}};   // gate f : tiles 0,1
    v8f acc1[2] = {{}, {}};   // gate i
    v8f acc2[2] = {{}, {}};   // gate c~
    v8f acc3[2] = {{}, {}};   // gate o

    for (int kk = 0; kk < H_SZ; kk += 32) {
        __builtin_prefetch(aBase0 + kk + 32, 0, 0);     // global_prefetch_b8
        __builtin_prefetch(aBase1 + kk + 32, 0, 0);

        v16bf a0, a1;
        const __bf16* ap0 = aBase0 + kk;
        const __bf16* ap1 = aBase1 + kk;
#pragma unroll
        for (int e = 0; e < 8; ++e) {
            a0[e] = ap0[e]; a0[e + 8] = ap0[e + 8];
            a1[e] = ap1[e]; a1[e + 8] = ap1[e + 8];
        }

        const v16bf b0 = *(const v16bf*)(bBase0 + kk);
        const v16bf b1 = *(const v16bf*)(bBase1 + kk);
        const v16bf b2 = *(const v16bf*)(bBase2 + kk);
        const v16bf b3 = *(const v16bf*)(bBase3 + kk);

        acc0[0] = __builtin_amdgcn_wmma_f32_16x16x32_bf16(false, a0, false, b0, (short)0, acc0[0], false, false);
        acc0[1] = __builtin_amdgcn_wmma_f32_16x16x32_bf16(false, a1, false, b0, (short)0, acc0[1], false, false);
        acc1[0] = __builtin_amdgcn_wmma_f32_16x16x32_bf16(false, a0, false, b1, (short)0, acc1[0], false, false);
        acc1[1] = __builtin_amdgcn_wmma_f32_16x16x32_bf16(false, a1, false, b1, (short)0, acc1[1], false, false);
        acc2[0] = __builtin_amdgcn_wmma_f32_16x16x32_bf16(false, a0, false, b2, (short)0, acc2[0], false, false);
        acc2[1] = __builtin_amdgcn_wmma_f32_16x16x32_bf16(false, a1, false, b2, (short)0, acc2[1], false, false);
        acc3[0] = __builtin_amdgcn_wmma_f32_16x16x32_bf16(false, a0, false, b3, (short)0, acc3[0], false, false);
        acc3[1] = __builtin_amdgcn_wmma_f32_16x16x32_bf16(false, a1, false, b3, (short)0, acc3[1], false, false);
    }

    // Epilogue: x @ Wg^T + bias for this lane's fixed column bcol
    float wf[IN_SZ], wi[IN_SZ], wc[IN_SZ], wo[IN_SZ];
#pragma unroll
    for (int k = 0; k < IN_SZ; ++k) {
        wf[k] = Wf[bcol * IN_SZ + k];
        wi[k] = Wi[bcol * IN_SZ + k];
        wc[k] = Wc[bcol * IN_SZ + k];
        wo[k] = Wo[bcol * IN_SZ + k];
    }
    const float biasf = bf_[bcol], biasi = bi_[bcol], biasc = bc_[bcol], biaso = bo_[bcol];

#pragma unroll
    for (int t = 0; t < 2; ++t) {
#pragma unroll
        for (int v = 0; v < 8; ++v) {
            const int row = row0 + t * 16 + v + hiHalf * 8;   // C/D row mapping
            const float* xr = x + (size_t)row * IN_SZ;
            float pf = acc0[t][v] + biasf;
            float pi = acc1[t][v] + biasi;
            float pc = acc2[t][v] + biasc;
            float po = acc3[t][v] + biaso;
#pragma unroll
            for (int k = 0; k < IN_SZ; ++k) {
                const float xv = xr[k];
                pf += xv * wf[k]; pi += xv * wi[k]; pc += xv * wc[k]; po += xv * wo[k];
            }
            const float f  = 1.f / (1.f + expf(-pf));
            const float ig = 1.f / (1.f + expf(-pi));
            const float ct = tanhf(pc);
            const float o  = 1.f / (1.f + expf(-po));
            const float cn = f * c_in[(size_t)row * H_SZ + bcol] + ig * ct;
            const float hs = o * tanhf(cn);

            float* rrow = rt + (size_t)row * SIXH + bcol;
            rrow[0 * H_SZ] = f;
            rrow[1 * H_SZ] = ig;
            rrow[2 * H_SZ] = ct;
            rrow[3 * H_SZ] = o;
            rrow[4 * H_SZ] = cn;
            rrow[5 * H_SZ] = hs;
            cnew_out[(size_t)row * H_SZ + bcol] = cn;
        }
    }
}

// ---------------------------------------------------------------------------
// GD: fused single pass per iteration. Block handles GD_ROWS rows:
// phase 1: per-row err = r_t[b,:]·theta - mp[b]   (wave-parallel dots)
// phase 2: grad[j] += sum_rows r_t[b,j]*err[b]    (one atomic per col per block)
// ---------------------------------------------------------------------------
__global__ __launch_bounds__(256)
void gd_acc_kernel(const float* __restrict__ rt, const float* __restrict__ mp,
                   const float* __restrict__ theta, float* __restrict__ grad)
{
    __shared__ float th_s[SIXH];
    __shared__ float errs[GD_ROWS];
    const int tid = threadIdx.x, lane = tid & 31, wave = tid >> 5;

    for (int j = tid; j < SIXH; j += 256) th_s[j] = theta[j];
    __syncthreads();

    const int rowBase = blockIdx.x * GD_ROWS;
    for (int rr = 0; rr < GD_ROWS / 8; ++rr) {
        const int r = wave * (GD_ROWS / 8) + rr;
        const int b = rowBase + r;
        const float* rrow = rt + (size_t)b * SIXH;
        float p = 0.f;
        for (int j = lane; j < SIXH; j += 32) p += rrow[j] * th_s[j];
#pragma unroll
        for (int off = 16; off > 0; off >>= 1) p += __shfl_xor(p, off, 32);
        if (lane == 0) errs[r] = p - mp[b];
    }
    __syncthreads();

    for (int j = tid; j < SIXH; j += 256) {
        float acc = 0.f;
#pragma unroll 4
        for (int r = 0; r < GD_ROWS; ++r)
            acc += rt[(size_t)(rowBase + r) * SIXH + j] * errs[r];
        atomicAdd(&grad[j], acc);
    }
}

__global__ void gd_init_kernel(float* theta, float* grad) {
    int i = blockIdx.x * blockDim.x + threadIdx.x;
    if (i < SIXH) { theta[i] = 0.f; grad[i] = 0.f; }
}

__global__ void gd_update_kernel(float* theta, float* grad) {
    int i = blockIdx.x * blockDim.x + threadIdx.x;
    if (i < SIXH) {
        theta[i] -= (2.f * GD_LR / (float)B_SZ) * grad[i];
        grad[i] = 0.f;   // ready for next iteration
    }
}

// ---------------------------------------------------------------------------
// importance = mean |theta| per block of 1024; argmax (first max wins)
// ---------------------------------------------------------------------------
__global__ void argmax_kernel(const float* __restrict__ theta, int* __restrict__ midx) {
    __shared__ float s[6][256];
    const int tid = threadIdx.x;
    float acc[6] = {0.f, 0.f, 0.f, 0.f, 0.f, 0.f};
    for (int j = tid; j < SIXH; j += 256) acc[j >> 10] += fabsf(theta[j]);
#pragma unroll
    for (int g = 0; g < 6; ++g) s[g][tid] = acc[g];
    __syncthreads();
    if (tid == 0) {
        float best = -1.f; int bi = 0;
        for (int g = 0; g < 6; ++g) {
            float t = 0.f;
            for (int k = 0; k < 256; ++k) t += s[g][k];
            if (t > best) { best = t; bi = g; }
        }
        *midx = bi;
    }
}

// ---------------------------------------------------------------------------
// h_new = r_t[:, midx*H : (midx+1)*H]
// ---------------------------------------------------------------------------
__global__ void slice_kernel(const float* __restrict__ rt, const int* __restrict__ midx,
                             float* __restrict__ hout) {
    const int mi = *midx;
    const size_t i4 = (size_t)blockIdx.x * blockDim.x + threadIdx.x;
    const size_t base = i4 * 4;
    const size_t b = base >> 10;
    const size_t j = base & 1023u;
    const float4 v = *(const float4*)(rt + b * SIXH + (size_t)mi * H_SZ + j);
    *(float4*)(hout + base) = v;
}

// ---------------------------------------------------------------------------
extern "C" void kernel_launch(void* const* d_in, const int* in_sizes, int n_in,
                              void* d_out, int out_size, void* d_ws, size_t ws_size,
                              hipStream_t stream)
{
    (void)in_sizes; (void)n_in; (void)out_size; (void)ws_size;

    const float* x   = (const float*)d_in[0];
    const float* hid = (const float*)d_in[1];
    const float* c   = (const float*)d_in[2];
    const float* mp  = (const float*)d_in[3];
    const float* Wf  = (const float*)d_in[4];
    const float* bf_ = (const float*)d_in[5];
    const float* Uf  = (const float*)d_in[6];
    const float* Wi  = (const float*)d_in[7];
    const float* bi_ = (const float*)d_in[8];
    const float* Ui  = (const float*)d_in[9];
    const float* Wc  = (const float*)d_in[10];
    const float* bc_ = (const float*)d_in[11];
    const float* Uc  = (const float*)d_in[12];
    const float* Wo  = (const float*)d_in[13];
    const float* bo_ = (const float*)d_in[14];
    const float* Uo  = (const float*)d_in[15];

    float* hout = (float*)d_out;                      // [B][H]
    float* cout = hout + (size_t)B_SZ * H_SZ;         // [B][H]

    char* w = (char*)d_ws;
    float*          rt    = (float*)w;           w += (size_t)B_SZ * SIXH * sizeof(float);
    unsigned short* hbf   = (unsigned short*)w;  w += (size_t)B_SZ * H_SZ * sizeof(unsigned short);
    unsigned short* ubf   = (unsigned short*)w;  w += (size_t)4 * H_SZ * H_SZ * sizeof(unsigned short);
    float*          theta = (float*)w;           w += SIXH * sizeof(float);
    float*          grad  = (float*)w;           w += SIXH * sizeof(float);
    int*            midx  = (int*)w;             w += 256;

    // 1) convert hidden + four U matrices to bf16 (vectorized by 4)
    const int nh4 = (B_SZ * H_SZ) / 4;
    const int nu  = H_SZ * H_SZ;
    const int nu4 = nu / 4;
    f2bf_kernel<<<(nh4 + 255) / 256, 256, 0, stream>>>(hid, hbf, nh4);
    f2bf_kernel<<<(nu4 + 255) / 256, 256, 0, stream>>>(Uf, ubf + 0 * (size_t)nu, nu4);
    f2bf_kernel<<<(nu4 + 255) / 256, 256, 0, stream>>>(Ui, ubf + 1 * (size_t)nu, nu4);
    f2bf_kernel<<<(nu4 + 255) / 256, 256, 0, stream>>>(Uc, ubf + 2 * (size_t)nu, nu4);
    f2bf_kernel<<<(nu4 + 255) / 256, 256, 0, stream>>>(Uo, ubf + 3 * (size_t)nu, nu4);

    // 2) fused gate GEMM + LSTM elementwise -> rt, c_new
    dim3 gg(B_SZ / 32, 8);
    gates_kernel<<<gg, 256, 0, stream>>>(hbf, ubf, x, c,
                                         Wf, bf_, Wi, bi_, Wc, bc_, Wo, bo_,
                                         rt, cout);

    // 3) gradient descent (7 fused passes)
    gd_init_kernel<<<(SIXH + 255) / 256, 256, 0, stream>>>(theta, grad);
    for (int it = 0; it < GD_ITERS; ++it) {
        gd_acc_kernel<<<B_SZ / GD_ROWS, 256, 0, stream>>>(rt, mp, theta, grad);
        gd_update_kernel<<<(SIXH + 255) / 256, 256, 0, stream>>>(theta, grad);
    }

    // 4) importance argmax + slice h_new
    argmax_kernel<<<1, 256, 0, stream>>>(theta, midx);
    slice_kernel<<<(B_SZ * (H_SZ / 4)) / 256, 256, 0, stream>>>(rt, midx, hout);
}
